// OdeBlock_33792802685607
// MI455X (gfx1250) — matrix operand order repaired and checked by reading, not compiled, not run on hardware
//
#include <hip/hip_runtime.h>
#include <hip/hip_bf16.h>
#include <math.h>

// ---------------------------------------------------------------------------
// dopri5 Neural-ODE on gfx1250 using v_wmma_f32_16x16x32_f16 matrix cores.
// State y: [8192, 72] f32.  f(y) = tanh(y@W1+b1)@W2+b2, W1:[72,256], W2:[256,72]
// 32 adaptive iterations; global RMS error reduction per step.
// ---------------------------------------------------------------------------

typedef _Float16 v16h __attribute__((ext_vector_type(16)));
typedef _Float16 v8h  __attribute__((ext_vector_type(8)));
typedef float    v8f  __attribute__((ext_vector_type(8)));

#define BATCH   8192
#define DA      72
#define HID     256
#define KP      96      // GEMM1 K padded to 3 chunks of 32
#define NP2     80      // GEMM2 N padded to 5 tiles of 16
#define NELEM   (BATCH * DA)     // 589824
#define TILE_E  (16 * DA)        // 1152 elements per 16-row tile
#define NBLK    (BATCH / 16)     // 512 step blocks
#define NITER   32

// Butcher tableau (float-folded)
__device__ constexpr float CA21 = (float)(1.0/5.0);
__device__ constexpr float CA31 = (float)(3.0/40.0),  CA32 = (float)(9.0/40.0);
__device__ constexpr float CA41 = (float)(44.0/45.0), CA42 = (float)(-56.0/15.0), CA43 = (float)(32.0/9.0);
__device__ constexpr float CA51 = (float)(19372.0/6561.0), CA52 = (float)(-25360.0/2187.0),
                           CA53 = (float)(64448.0/6561.0), CA54 = (float)(-212.0/729.0);
__device__ constexpr float CA61 = (float)(9017.0/3168.0),  CA62 = (float)(-355.0/33.0),
                           CA63 = (float)(46732.0/5247.0), CA64 = (float)(49.0/176.0),
                           CA65 = (float)(-5103.0/18656.0);
__device__ constexpr float CB1 = (float)(35.0/384.0), CB3 = (float)(500.0/1113.0),
                           CB4 = (float)(125.0/192.0), CB5 = (float)(-2187.0/6784.0),
                           CB6 = (float)(11.0/84.0);
__device__ constexpr float CE1 = (float)(35.0/384.0 - 5179.0/57600.0);
__device__ constexpr float CE3 = (float)(500.0/1113.0 - 7571.0/16695.0);
__device__ constexpr float CE4 = (float)(125.0/192.0 - 393.0/640.0);
__device__ constexpr float CE5 = (float)(-2187.0/6784.0 + 92097.0/339200.0);
__device__ constexpr float CE6 = (float)(11.0/84.0 - 187.0/2100.0);
__device__ constexpr float CE7 = (float)(-1.0/40.0);

// Branch-free tanh. Prefer the gfx1250 hardware V_TANH_F32; otherwise a single
// v_exp_f32 based formulation with no exec divergence (overflow -> inf -> +-1).
__device__ __forceinline__ float fast_tanh(float x) {
#if __has_builtin(__builtin_amdgcn_tanhf)
  return __builtin_amdgcn_tanhf(x);
#else
  const float ax = fabsf(x);
  const float e  = __expf(2.0f * ax);      // v_exp_f32 path, branchless
  const float t  = 1.0f - 2.0f / (e + 1.0f);
  return copysignf(t, x);
#endif
}

// A-fragment (16x32 f16): lane (l<16) row=l holds K {0..7,16..23}; lane l+16 holds {8..15,24..31}
__device__ __forceinline__ v16h load_a_frag(const _Float16* p) {
  v8h lo = *(const v8h*)(p);       // K j .. j+7
  v8h hh = *(const v8h*)(p + 16);  // K j+16 .. j+23
  v16h r;
#pragma unroll
  for (int i = 0; i < 8; ++i) { r[i] = lo[i]; r[i + 8] = hh[i]; }
  return r;
}

// f(X) for one 16-row tile. Xh: [16][KP] f16 (cols >= DA zero), Hh: [16][HID] f16 scratch,
// kout: [16][DA] f32. Weights pre-transposed: W1t[n][k] (n<HID, k<KP), W2t[n][k] (n<NP2, k<HID).
__device__ __forceinline__ void feval(const _Float16* Xh, _Float16* Hh, float* kout,
                                      const _Float16* __restrict__ W1t,
                                      const _Float16* __restrict__ W2t,
                                      const float* __restrict__ b1v,
                                      const float* __restrict__ b2v,
                                      int lane)
{
  const int mr = lane & 15;
  const int hi = lane >> 4;
  __syncthreads();

  // ---- GEMM1: H = tanh(X[16xKP] @ W1[KPxHID] + b1) ----
  v16h a1[3];
#pragma unroll
  for (int kc = 0; kc < 3; ++kc)
    a1[kc] = load_a_frag(Xh + mr * KP + kc * 32 + hi * 8);

  for (int t = 0; t < 16; ++t) {
    v8f acc = {};
    const int n = t * 16 + mr;            // output column owned by this lane
    const _Float16* wrow = W1t + n * KP;  // B frag: lane column n, K {8h..8h+7, 16+8h..16+8h+7}
#pragma unroll
    for (int kc = 0; kc < 3; ++kc) {
      v16h b = load_a_frag(wrow + kc * 32 + hi * 8);
      acc = __builtin_amdgcn_wmma_f32_16x16x32_f16(false, a1[kc], false, b,
                                                   (short)0, acc, false, false);
    }
    const float bias = b1v[n];
#pragma unroll
    for (int i = 0; i < 8; ++i)           // D: rows hi*8+i, column n
      Hh[(hi * 8 + i) * HID + n] = (_Float16)fast_tanh(acc[i] + bias);
  }
  __syncthreads();

  // ---- GEMM2: out = H[16xHID] @ W2[HIDxDA] + b2 ----
  v16h a2[8];
#pragma unroll
  for (int kc = 0; kc < 8; ++kc)
    a2[kc] = load_a_frag(Hh + mr * HID + kc * 32 + hi * 8);

  for (int t = 0; t < 5; ++t) {
    v8f acc = {};
    const int n = t * 16 + mr;
    const _Float16* wrow = W2t + n * HID;
#pragma unroll
    for (int kc = 0; kc < 8; ++kc) {
      v16h b = load_a_frag(wrow + kc * 32 + hi * 8);
      acc = __builtin_amdgcn_wmma_f32_16x16x32_f16(false, a2[kc], false, b,
                                                   (short)0, acc, false, false);
    }
    if (n < DA) {                         // drop padded columns 72..79
      const float bias = b2v[n];
#pragma unroll
      for (int i = 0; i < 8; ++i)
        kout[(hi * 8 + i) * DA + n] = acc[i] + bias;
    }
  }
  __syncthreads();
}

// One dopri5 step for a 16-row tile: compute k1..k7, y5, squared-error partial.
__global__ void __launch_bounds__(32)
ode_step_kernel(const float* __restrict__ y, float* __restrict__ y5g,
                const _Float16* __restrict__ W1t, const _Float16* __restrict__ W2t,
                const float* __restrict__ b1v, const float* __restrict__ b2v,
                const float* __restrict__ scal, float* __restrict__ partials)
{
  __shared__ __align__(32) _Float16 Xh[16 * KP];
  __shared__ __align__(32) _Float16 Hh[16 * HID];
  __shared__ float ks[7][TILE_E];
  __shared__ float yt[TILE_E];
  __shared__ float y5t[TILE_E];

  const int lane = threadIdx.x;
  const int base = blockIdx.x * TILE_E;

  const float t  = scal[0];
  const float dt = scal[1];
  const bool done = (t >= 0.9f - 1e-7f);
  const float h = done ? dt : fminf(dt, 0.9f - t);

  // zero Xh once; stage writes only touch cols < DA, padding stays 0
  for (int e = lane; e < 16 * KP; e += 32) Xh[e] = (_Float16)0.0f;

  // stage 1: X = y
  for (int e = lane; e < TILE_E; e += 32) {
    const float v = y[base + e];
    yt[e] = v;
    const int m = e / DA, c = e - m * DA;
    Xh[m * KP + c] = (_Float16)v;
  }
  feval(Xh, Hh, ks[0], W1t, W2t, b1v, b2v, lane);               // k1

  for (int e = lane; e < TILE_E; e += 32) {
    const float v = yt[e] + h * (CA21 * ks[0][e]);
    const int m = e / DA, c = e - m * DA;
    Xh[m * KP + c] = (_Float16)v;
  }
  feval(Xh, Hh, ks[1], W1t, W2t, b1v, b2v, lane);               // k2

  for (int e = lane; e < TILE_E; e += 32) {
    const float v = yt[e] + h * (CA31 * ks[0][e] + CA32 * ks[1][e]);
    const int m = e / DA, c = e - m * DA;
    Xh[m * KP + c] = (_Float16)v;
  }
  feval(Xh, Hh, ks[2], W1t, W2t, b1v, b2v, lane);               // k3

  for (int e = lane; e < TILE_E; e += 32) {
    const float v = yt[e] + h * (CA41 * ks[0][e] + CA42 * ks[1][e] + CA43 * ks[2][e]);
    const int m = e / DA, c = e - m * DA;
    Xh[m * KP + c] = (_Float16)v;
  }
  feval(Xh, Hh, ks[3], W1t, W2t, b1v, b2v, lane);               // k4

  for (int e = lane; e < TILE_E; e += 32) {
    const float v = yt[e] + h * (CA51 * ks[0][e] + CA52 * ks[1][e] +
                                 CA53 * ks[2][e] + CA54 * ks[3][e]);
    const int m = e / DA, c = e - m * DA;
    Xh[m * KP + c] = (_Float16)v;
  }
  feval(Xh, Hh, ks[4], W1t, W2t, b1v, b2v, lane);               // k5

  for (int e = lane; e < TILE_E; e += 32) {
    const float v = yt[e] + h * (CA61 * ks[0][e] + CA62 * ks[1][e] + CA63 * ks[2][e] +
                                 CA64 * ks[3][e] + CA65 * ks[4][e]);
    const int m = e / DA, c = e - m * DA;
    Xh[m * KP + c] = (_Float16)v;
  }
  feval(Xh, Hh, ks[5], W1t, W2t, b1v, b2v, lane);               // k6

  // y5 (5th-order solution); also FSAL input for k7
  for (int e = lane; e < TILE_E; e += 32) {
    const float v = yt[e] + h * (CB1 * ks[0][e] + CB3 * ks[2][e] + CB4 * ks[3][e] +
                                 CB5 * ks[4][e] + CB6 * ks[5][e]);
    y5t[e] = v;
    y5g[base + e] = v;
    const int m = e / DA, c = e - m * DA;
    Xh[m * KP + c] = (_Float16)v;
  }
  feval(Xh, Hh, ks[6], W1t, W2t, b1v, b2v, lane);               // k7

  // squared-error partial for this tile
  float s = 0.0f;
  for (int e = lane; e < TILE_E; e += 32) {
    const float ev = h * (CE1 * ks[0][e] + CE3 * ks[2][e] + CE4 * ks[3][e] +
                          CE5 * ks[4][e] + CE6 * ks[5][e] + CE7 * ks[6][e]);
    const float sc = 1e-3f + 1e-3f * fmaxf(fabsf(yt[e]), fabsf(y5t[e]));
    const float r = ev / sc;
    s += r * r;
  }
#pragma unroll
  for (int off = 16; off > 0; off >>= 1) s += __shfl_xor(s, off, 32);
  if (lane == 0) partials[blockIdx.x] = s;
}

// Deterministic scalar update: err -> accept, t, dt
__global__ void update_kernel(float* __restrict__ scal, const float* __restrict__ part,
                              int* __restrict__ accept)
{
  float sum = 0.0f;
  for (int i = 0; i < NBLK; ++i) sum += part[i];
  float err = sqrtf(sum / (float)NELEM);
  err = fmaxf(err, 1e-10f);
  const float t = scal[0], dt = scal[1];
  const bool done = (t >= 0.9f - 1e-7f);
  const float h = done ? dt : fminf(dt, 0.9f - t);
  const bool acc = (err <= 1.0f) && (!done);
  if (acc) scal[0] = t + h;
  float factor = 0.9f * powf(err, -0.2f);
  factor = fminf(fmaxf(factor, 0.2f), 10.0f);
  if (!done) scal[1] = dt * factor;
  *accept = acc ? 1 : 0;
}

__global__ void apply_kernel(float* __restrict__ y, const float* __restrict__ y5,
                             const int* __restrict__ accept)
{
  if (*accept == 0) return;
  const int i = blockIdx.x * blockDim.x + threadIdx.x;
  if (i < NELEM) y[i] = y5[i];
}

__global__ void init_y_kernel(const float* __restrict__ x, float* __restrict__ yb)
{
  const int i = blockIdx.x * blockDim.x + threadIdx.x;
  if (i < NELEM) {
    const int m = i / DA, c = i - m * DA;
    yb[i] = (c < 64) ? x[m * 64 + c] : 0.0f;   // augment with 8 zero dims
  }
}

// Transposed, K/N-padded f16 weights: W1t[n][k]=W1[k][n], W2t[n][k]=W2[k][n]
__global__ void init_w_kernel(const float* __restrict__ W1, const float* __restrict__ W2,
                              _Float16* __restrict__ W1t, _Float16* __restrict__ W2t)
{
  const int i = blockIdx.x * blockDim.x + threadIdx.x;
  if (i < HID * KP) {
    const int n = i / KP, k = i - n * KP;
    W1t[i] = (k < DA) ? (_Float16)W1[k * HID + n] : (_Float16)0.0f;
  } else if (i < HID * KP + NP2 * HID) {
    const int j = i - HID * KP;
    const int n = j / HID, k = j - n * HID;
    W2t[j] = (n < DA) ? (_Float16)W2[k * DA + n] : (_Float16)0.0f;
  }
}

__global__ void init_scal_kernel(float* __restrict__ scal, int* __restrict__ accept)
{
  scal[0] = 0.0f;   // t
  scal[1] = 0.09f;  // dt
  *accept = 0;
}

__global__ void copy_out_kernel(const float* __restrict__ y, float* __restrict__ out)
{
  const int i = blockIdx.x * blockDim.x + threadIdx.x;
  if (i < NELEM) out[i] = y[i];
}

extern "C" void kernel_launch(void* const* d_in, const int* in_sizes, int n_in,
                              void* d_out, int out_size, void* d_ws, size_t ws_size,
                              hipStream_t stream)
{
  const float* x  = (const float*)d_in[0];  // [8192,64]
  const float* W1 = (const float*)d_in[1];  // [72,256]
  const float* b1 = (const float*)d_in[2];  // [256]
  const float* W2 = (const float*)d_in[3];  // [256,72]
  const float* b2 = (const float*)d_in[4];  // [72]
  (void)in_sizes; (void)n_in; (void)out_size; (void)ws_size;

  char* w = (char*)d_ws;
  size_t off = 0;
  auto alloc = [&](size_t bytes) -> void* {
    void* p = (void*)(w + off);
    off += (bytes + 255) & ~(size_t)255;
    return p;
  };
  float*    yb   = (float*)alloc((size_t)NELEM * 4);
  float*    y5b  = (float*)alloc((size_t)NELEM * 4);
  _Float16* W1t  = (_Float16*)alloc((size_t)HID * KP * 2);
  _Float16* W2t  = (_Float16*)alloc((size_t)NP2 * HID * 2);
  float*    part = (float*)alloc((size_t)NBLK * 4);
  float*    scal = (float*)alloc(2 * 4);
  int*      acc  = (int*)alloc(4);

  init_y_kernel<<<(NELEM + 255) / 256, 256, 0, stream>>>(x, yb);
  init_w_kernel<<<(HID * KP + NP2 * HID + 255) / 256, 256, 0, stream>>>(W1, W2, W1t, W2t);
  init_scal_kernel<<<1, 1, 0, stream>>>(scal, acc);

  for (int it = 0; it < NITER; ++it) {
    ode_step_kernel<<<NBLK, 32, 0, stream>>>(yb, y5b, W1t, W2t, b1, b2, scal, part);
    update_kernel<<<1, 1, 0, stream>>>(scal, part, acc);
    apply_kernel<<<(NELEM + 255) / 256, 256, 0, stream>>>(yb, y5b, acc);
  }

  copy_out_kernel<<<(NELEM + 255) / 256, 256, 0, stream>>>(yb, (float*)d_out);
}
